// StyleGuidedAttention_3607772528745
// MI455X (gfx1250) — matrix operand order repaired and checked by reading, not compiled, not run on hardware
//
#include <hip/hip_runtime.h>
#include <hip/hip_bf16.h>
#include <math.h>

typedef __attribute__((ext_vector_type(16))) _Float16 v16h;
typedef __attribute__((ext_vector_type(8)))  _Float16 v8h;
typedef __attribute__((ext_vector_type(8)))  float    v8f;
typedef __attribute__((ext_vector_type(4)))  float    v4f;

#define B_    128
#define N_    1369
#define DINO_ 384
#define CLIP_ 512
#define H_    8
#define D_    64
#define EPS_  1e-5f

// LDS weight layout: [sel in {K,V}][col 0..63][k 0..383] halves, k-stride padded
#define KPAD     392                     // 384 + 8 halves -> 784B col stride, bank-spread
#define WSTRIDE  (D_ * KPAD)             // halves per matrix
#define DYN_LDS_BYTES (2 * WSTRIDE * 2)  // 100,352 bytes

union HF { v16h v; v8h h[2]; };

// ---------------------------------------------------------------------------
// Kernel 1: Q = clip @ Wq + bq   (128 x 512 @ 512x512, trivial)
// ---------------------------------------------------------------------------
__global__ __launch_bounds__(512)
void qproj_kernel(const float* __restrict__ clip,
                  const float* __restrict__ Wq,
                  const float* __restrict__ bq,
                  float* __restrict__ qout) {
  __shared__ float xs[CLIP_];
  const int b = blockIdx.x;
  const int c = threadIdx.x;              // 512 threads = 512 output cols
  xs[c] = clip[(size_t)b * CLIP_ + c];
  __syncthreads();
  float acc = bq[c];
  #pragma unroll 8
  for (int k = 0; k < CLIP_; ++k)
    acc = fmaf(xs[k], Wq[(size_t)k * CLIP_ + c], acc);
  qout[(size_t)b * CLIP_ + c] = acc;
}

// ---------------------------------------------------------------------------
// Kernel 2: fused K/V projection + flash attention per (batch, head).
// grid = B*H blocks, 256 threads (8 waves of 32). Each wave owns 2 x 16
// patches per tile (B fragments feed two WMMAs); 256 patches per workgroup
// tile; 6 tiles cover N=1369.
// ---------------------------------------------------------------------------
__global__ __launch_bounds__(256)
void fused_attn_kernel(const float* __restrict__ dino,
                       const float* __restrict__ Wk,
                       const float* __restrict__ Wv,
                       const float* __restrict__ bv,
                       const float* __restrict__ qvec,
                       const float* __restrict__ temperature,
                       float* __restrict__ attn_out) {
  extern __shared__ _Float16 wsmem[];     // f16 weights, [2][64][KPAD]
  __shared__ float red[8][66];            // per-wave merge: 64 cols + m + l

  const int bh   = blockIdx.x;
  const int b    = bh >> 3;
  const int h    = bh & 7;
  const int tid  = threadIdx.x;
  const int wave = tid >> 5;
  const int lane = tid & 31;
  const int l16  = lane & 15;
  const bool hiHalf = lane >= 16;

  // ---- stage this head's Wk/Wv slices into LDS as f16 -------------------
  for (int idx = tid; idx < 2 * D_ * DINO_; idx += 256) {
    const int col = idx & (D_ - 1);
    const int k   = (idx >> 6) % DINO_;
    const int sel = idx / (D_ * DINO_);
    const float* W = sel ? Wv : Wk;
    const float w = W[(size_t)k * CLIP_ + h * D_ + col];
    wsmem[sel * WSTRIDE + col * KPAD + k] = (_Float16)w;
  }
  __syncthreads();

  // ---- per-lane Q, pre-divided by sqrt(D)*temperature -------------------
  const float scale = 8.0f * temperature[0];
  float q[4];
  #pragma unroll
  for (int ct = 0; ct < 4; ++ct)
    q[ct] = qvec[(size_t)b * CLIP_ + h * D_ + ct * 16 + l16] / scale;

  const float* Ab = dino + (size_t)b * N_ * DINO_;

  float m_run = -__builtin_inff();
  float l_run = 0.0f;
  float outc[4] = {0.f, 0.f, 0.f, 0.f};

  const int NITER = (N_ + 255) / 256;     // 6
  for (int it = 0; it < NITER; ++it) {
    const int base = it * 256 + wave * 32;  // first patch of this wave's 32

    v8f accK[2][4], accV[2][4];
    #pragma unroll
    for (int t = 0; t < 2; ++t)
      #pragma unroll
      for (int ct = 0; ct < 4; ++ct) {
        accK[t][ct] = (v8f){0, 0, 0, 0, 0, 0, 0, 0};
        accV[t][ct] = (v8f){0, 0, 0, 0, 0, 0, 0, 0};
      }

    // A-layout rows for this lane (lanes 0-15 and 16-31 both map M=0..15)
    const float* Arow[2];
    #pragma unroll
    for (int t = 0; t < 2; ++t) {
      const int rr = base + t * 16 + l16;
      const int rc = (rr < N_) ? rr : (N_ - 1);   // clamp OOB
      Arow[t] = Ab + (size_t)rc * DINO_;
    }

    for (int ks = 0; ks < DINO_ / 32; ++ks) {
      // A fragments (16x32 f16 each): lanes<16 hold K ks*32+0..15,
      // lanes>=16 hold K ks*32+16..31.
      const int kb = ks * 32 + (hiHalf ? 16 : 0);
      v16h af[2];
      #pragma unroll
      for (int t = 0; t < 2; ++t) {
        const v4f r0 = *(const v4f*)(Arow[t] + kb);
        const v4f r1 = *(const v4f*)(Arow[t] + kb + 4);
        const v4f r2 = *(const v4f*)(Arow[t] + kb + 8);
        const v4f r3 = *(const v4f*)(Arow[t] + kb + 12);
        #pragma unroll
        for (int i = 0; i < 4; ++i) {
          af[t][i]      = (_Float16)r0[i];
          af[t][4 + i]  = (_Float16)r1[i];
          af[t][8 + i]  = (_Float16)r2[i];
          af[t][12 + i] = (_Float16)r3[i];
        }
      }

      // B fragments: lane col = ct*16 + l16, K run = ks*32 + (hi?16:0) + 0..15
      const int kofs = ks * 32 + (hiHalf ? 16 : 0);

      // --- K-weight group: 8 ds_load_b128 in flight, then 8 WMMAs ---------
      HF bk[4];
      #pragma unroll
      for (int ct = 0; ct < 4; ++ct) {
        const _Float16* wp = &wsmem[(ct * 16 + l16) * KPAD + kofs];
        bk[ct].h[0] = *(const v8h*)(wp);
        bk[ct].h[1] = *(const v8h*)(wp + 8);
      }
      #pragma unroll
      for (int ct = 0; ct < 4; ++ct)
        #pragma unroll
        for (int t = 0; t < 2; ++t)
          accK[t][ct] = __builtin_amdgcn_wmma_f32_16x16x32_f16(
              false, af[t], false, bk[ct].v, (short)0, accK[t][ct], false, false);

      // --- V-weight group -------------------------------------------------
      HF bw[4];
      #pragma unroll
      for (int ct = 0; ct < 4; ++ct) {
        const _Float16* wp = &wsmem[WSTRIDE + (ct * 16 + l16) * KPAD + kofs];
        bw[ct].h[0] = *(const v8h*)(wp);
        bw[ct].h[1] = *(const v8h*)(wp + 8);
      }
      #pragma unroll
      for (int ct = 0; ct < 4; ++ct)
        #pragma unroll
        for (int t = 0; t < 2; ++t)
          accV[t][ct] = __builtin_amdgcn_wmma_f32_16x16x32_f16(
              false, af[t], false, bw[ct].v, (short)0, accV[t][ct], false, false);
    }

    // ---- logits: lr[t][r] = Q . K[patch], patch = base + t*16 + r + 8*hi --
    float lr[2][8];
    #pragma unroll
    for (int t = 0; t < 2; ++t)
      #pragma unroll
      for (int r = 0; r < 8; ++r) {
        float p = q[0] * accK[t][0][r] + q[1] * accK[t][1][r] +
                  q[2] * accK[t][2][r] + q[3] * accK[t][3][r];
        p += __shfl_xor(p, 1, 16);
        p += __shfl_xor(p, 2, 16);
        p += __shfl_xor(p, 4, 16);
        p += __shfl_xor(p, 8, 16);
        const int patch = base + t * 16 + r + (hiHalf ? 8 : 0);
        lr[t][r] = (patch < N_) ? p : -__builtin_inff();
      }

    // ---- online softmax update -------------------------------------------
    float tmax = lr[0][0];
    #pragma unroll
    for (int t = 0; t < 2; ++t)
      #pragma unroll
      for (int r = 0; r < 8; ++r) tmax = fmaxf(tmax, lr[t][r]);
    tmax = fmaxf(tmax, __shfl_xor(tmax, 16, 32));
    const float m_new = fmaxf(m_run, tmax);
    if (m_new > -__builtin_inff()) {        // wave-uniform
      const float corr = __expf(m_run - m_new);
      float pr[2][8];
      float psum = 0.f;
      #pragma unroll
      for (int t = 0; t < 2; ++t)
        #pragma unroll
        for (int r = 0; r < 8; ++r) {
          pr[t][r] = __expf(lr[t][r] - m_new);
          psum += pr[t][r];
        }
      psum += __shfl_xor(psum, 16, 32);     // other half's patches
      l_run = l_run * corr + psum;
      #pragma unroll
      for (int ct = 0; ct < 4; ++ct) {
        float o = outc[ct] * corr;
        #pragma unroll
        for (int t = 0; t < 2; ++t)
          #pragma unroll
          for (int r = 0; r < 8; ++r) o = fmaf(pr[t][r], accV[t][ct][r], o);
        outc[ct] = o;
      }
      m_run = m_new;
    }
  }

  // ---- merge the two lane-halves (same cols, disjoint patches) ----------
  #pragma unroll
  for (int ct = 0; ct < 4; ++ct) outc[ct] += __shfl_xor(outc[ct], 16, 32);

  if (lane < 16) {
    #pragma unroll
    for (int ct = 0; ct < 4; ++ct) red[wave][ct * 16 + lane] = outc[ct];
  }
  if (lane == 0) { red[wave][64] = m_run; red[wave][65] = l_run; }
  __syncthreads();

  // ---- merge 8 waves (flash split-merge), add bv, write head slice ------
  if (tid < D_) {
    float m = red[0][64];
    #pragma unroll
    for (int w = 1; w < 8; ++w) m = fmaxf(m, red[w][64]);
    float num = 0.f, den = 0.f;
    #pragma unroll
    for (int w = 0; w < 8; ++w) {
      const float s = __expf(red[w][64] - m);
      num += s * red[w][tid];
      den += s * red[w][65];
    }
    attn_out[(size_t)b * CLIP_ + h * D_ + tid] = num / den + bv[h * D_ + tid];
  }
}

// ---------------------------------------------------------------------------
// Kernel 3: LayerNorm over last dim (512) per batch row
// ---------------------------------------------------------------------------
__global__ __launch_bounds__(256)
void layernorm_kernel(const float* __restrict__ x,
                      const float* __restrict__ gamma,
                      const float* __restrict__ beta,
                      float* __restrict__ out) {
  __shared__ float s1[256], s2[256];
  const int b = blockIdx.x, t = threadIdx.x;
  const float x0 = x[(size_t)b * 512 + t];
  const float x1 = x[(size_t)b * 512 + 256 + t];
  s1[t] = x0 + x1;
  s2[t] = x0 * x0 + x1 * x1;
  __syncthreads();
  for (int off = 128; off > 0; off >>= 1) {
    if (t < off) { s1[t] += s1[t + off]; s2[t] += s2[t + off]; }
    __syncthreads();
  }
  const float mu  = s1[0] * (1.0f / 512.0f);
  const float var = s2[0] * (1.0f / 512.0f) - mu * mu;
  const float inv = rsqrtf(var + EPS_);
  out[(size_t)b * 512 + t]       = (x0 - mu) * inv * gamma[t]       + beta[t];
  out[(size_t)b * 512 + 256 + t] = (x1 - mu) * inv * gamma[256 + t] + beta[256 + t];
}

// ---------------------------------------------------------------------------
extern "C" void kernel_launch(void* const* d_in, const int* in_sizes, int n_in,
                              void* d_out, int out_size, void* d_ws, size_t ws_size,
                              hipStream_t stream) {
  (void)in_sizes; (void)n_in; (void)out_size; (void)ws_size;
  const float* dino  = (const float*)d_in[0];
  const float* clip  = (const float*)d_in[1];
  const float* Wq    = (const float*)d_in[2];
  const float* bq    = (const float*)d_in[3];
  const float* Wk    = (const float*)d_in[4];
  // d_in[5] = bk: constant logit shift, exactly cancels in softmax
  const float* Wv    = (const float*)d_in[6];
  const float* bv    = (const float*)d_in[7];
  const float* temp  = (const float*)d_in[8];
  const float* gamma = (const float*)d_in[9];
  const float* beta  = (const float*)d_in[10];

  float* qws = (float*)d_ws;               // B*512 floats
  float* aws = qws + (size_t)B_ * CLIP_;   // B*512 floats

  qproj_kernel<<<B_, CLIP_, 0, stream>>>(clip, Wq, bq, qws);
  fused_attn_kernel<<<B_ * H_, 256, DYN_LDS_BYTES, stream>>>(
      dino, Wk, Wv, bv, qws, temp, aws);
  layernorm_kernel<<<B_, 256, 0, stream>>>(aws, gamma, beta, (float*)d_out);
}
